// CRFLayer_12567074308635
// MI455X (gfx1250) — matrix operand order repaired and checked by reading, not compiled, not run on hardware
//
#include <hip/hip_runtime.h>

typedef __attribute__((ext_vector_type(16))) _Float16 v16h;
typedef __attribute__((ext_vector_type(8)))  float    v8f;

#define S_LEN 1024
#define KT    64
#define NB    16            // batches per workgroup (WMMA M-tile)
#define NWAVE 16
#define WGSZ  (NWAVE * 32)  // 512 threads = 16 wave32

// ---------------------------------------------------------------------------
// Prep: Tw[i][j][k] = (f16) T[k][i][j]   (B-fragment friendly layout)
// ---------------------------------------------------------------------------
__global__ __launch_bounds__(WGSZ)
void crf2_prep_T(const float* __restrict__ T, _Float16* __restrict__ Tw) {
    int o = blockIdx.x * blockDim.x + threadIdx.x;
    if (o < KT * KT * KT) {
        int k = o & 63, j = (o >> 6) & 63, i = o >> 12;
        Tw[o] = (_Float16)T[(k * KT + i) * KT + j];
    }
}

// ---------------------------------------------------------------------------
// Scan kernel: one WG per 16 batches; state G[b][i][k] (=A[b,k,i]) in LDS f16.
// Per step, for each i:  Out_i (16b x 64j) = G[:,i,:] (16x64) @ T[:,i,:] (64x64)
// done as 4 j-tiles x 2 K-halves of v_wmma_f32_16x16x32_f16.
// ---------------------------------------------------------------------------
__global__ __launch_bounds__(WGSZ)
void crf2_scan(const float* __restrict__ X, const _Float16* __restrict__ Tw,
               float* __restrict__ Y) {
    extern __shared__ char lds[];
    _Float16* G0   = (_Float16*)lds;              // 128 KB
    _Float16* G1   = G0 + NB * KT * KT;           // 128 KB
    float*    xbuf = (float*)(G1 + NB * KT * KT); // 4 KB
    float*    ybuf = xbuf + NB * KT;              // 4 KB

    const int  tid   = threadIdx.x;
    const int  wave  = tid >> 5;
    const int  lane  = tid & 31;
    const int  hi    = lane >> 4;        // 0: low half lanes, 1: high half
    const int  nn    = lane & 15;        // row (A) / column (B,D) index in tile
    const int  mbase = hi * 8;           // this lane's D-row block
    const int  jt    = wave & 3;         // j-tile 0..3
    const int  i0    = (wave >> 2) * 16; // this wave's i range
    const int  jbase = jt * 16;
    const long b0    = (long)blockIdx.x * NB;

    // ---- t = 0: y[:,0,:] = x0 ; G[b,i,k] = x0[b,i] / KT ----
    for (int idx = tid; idx < NB * KT; idx += WGSZ) {
        int b = idx >> 6, j = idx & 63;
        float v = X[(b0 + b) * (S_LEN * KT) + j];
        xbuf[idx] = v;
        Y[(b0 + b) * (S_LEN * KT) + j] = v;
    }
    __syncthreads();
    for (int idx = tid; idx < NB * KT * KT; idx += WGSZ) {
        int i = (idx >> 6) & 63, b = idx >> 12;
        G0[idx] = (_Float16)(xbuf[b * KT + i] * (1.0f / KT));
    }
    __syncthreads();

    _Float16* Gold = G0;
    _Float16* Gnew = G1;

    for (int t = 1; t < S_LEN; ++t) {
        // phase 1: stage x_t, zero y accumulator
        for (int idx = tid; idx < NB * KT; idx += WGSZ) {
            int b = idx >> 6, j = idx & 63;
            xbuf[idx] = X[((b0 + b) * S_LEN + t) * KT + j];
            ybuf[idx] = 0.0f;
        }
        __syncthreads();

        // per-lane x values for its 8 D rows: x[mbase+v][jbase+nn]
        float xcol[8];
        #pragma unroll
        for (int v = 0; v < 8; ++v)
            xcol[v] = xbuf[(mbase + v) * KT + jbase + nn];

        float yp[8];
        #pragma unroll
        for (int v = 0; v < 8; ++v) yp[v] = 0.0f;

        // scatter base for this lane into G_new: row j = jbase+nn, col = i
        _Float16* gnb = Gnew + ((mbase * KT + jbase + nn) * KT) + i0;

        #pragma unroll 2
        for (int ii = 0; ii < 16; ++ii) {
            const int i = i0 + ii;

            // B fragments (L2-resident Tw): 16 contiguous halves per lane.
            // layout: lanes 0-15 -> K=0..15, lanes 16-31 -> K=16..31
            const _Float16* bptr = Tw + ((i * KT + jbase + nn) * KT) + hi * 16;
            if (ii + 1 < 16)
                __builtin_prefetch(bptr + KT * KT, 0, 1); // next i-slice
            v16h bf0 = *(const v16h*)(bptr);              // K-half 0
            v16h bf1 = *(const v16h*)(bptr + 32);         // K-half 1

            // A fragments from LDS state G[b=nn][i][k] (contiguous in k).
            // A k order chosen here (not the ISA table order): lane<16 holds K 0..15, lane>=16 K 16..31 (matches B)
            union { v16h v; uint4 q[2]; } a0, a1;
            const uint4* ap = (const uint4*)(Gold + ((nn * KT + i) * KT) + hi * 16);
            a0.q[0] = ap[0];  // k = hi*16     .. +7
            a0.q[1] = ap[1];  // k = hi*16 + 8 .. +15
            a1.q[0] = ap[4];  // +32
            a1.q[1] = ap[5];  // +40

            v8f c = {};
            c = __builtin_amdgcn_wmma_f32_16x16x32_f16(false, a0.v, false, bf0,
                                                       (short)0, c, false, false);
            c = __builtin_amdgcn_wmma_f32_16x16x32_f16(false, a1.v, false, bf1,
                                                       (short)0, c, false, false);

            // A_new = x * Out ; accumulate y ; scatter transposed into G_new
            #pragma unroll
            for (int v = 0; v < 8; ++v) {
                float val = c[v] * xcol[v];
                yp[v] += val;
                gnb[v * KT * KT + ii] = (_Float16)val;   // const offsets off base
            }
        }

        // reduce y partials across the 4 waves per j-tile (ds_add_f32)
        float* ybl = ybuf + mbase * KT + jbase + nn;
        #pragma unroll
        for (int v = 0; v < 8; ++v)
            atomicAdd(ybl + v * KT, yp[v]);
        __syncthreads();

        // phase 4: y -> global (coalesced per batch row)
        for (int idx = tid; idx < NB * KT; idx += WGSZ) {
            int b = idx >> 6, j = idx & 63;
            Y[((b0 + b) * S_LEN + t) * KT + j] = ybuf[idx];
        }
        __syncthreads();

        _Float16* tmp = Gold; Gold = Gnew; Gnew = tmp;
    }
}

// ---------------------------------------------------------------------------
extern "C" void kernel_launch(void* const* d_in, const int* in_sizes, int n_in,
                              void* d_out, int out_size, void* d_ws, size_t ws_size,
                              hipStream_t stream) {
    const float* X = (const float*)d_in[0];   // (64, 1024, 64) f32
    const float* T = (const float*)d_in[1];   // (64, 64, 64)  f32
    float*       Y = (float*)d_out;           // (64, 1024, 64) f32
    _Float16*    Tw = (_Float16*)d_ws;        // 512 KB converted/transposed T

    const int nT = KT * KT * KT;
    crf2_prep_T<<<(nT + WGSZ - 1) / WGSZ, WGSZ, 0, stream>>>(T, Tw);

    const size_t ldsBytes = (size_t)2 * NB * KT * KT * sizeof(_Float16)
                          + (size_t)2 * NB * KT * sizeof(float); // 264 KB
    crf2_scan<<<64 / NB, WGSZ, ldsBytes, stream>>>(X, Tw, Y);
}
